// NLGNN_15075335209141
// MI455X (gfx1250) — matrix operand order repaired and checked by reading, not compiled, not run on hardware
//
#include <hip/hip_runtime.h>
#include <hip/hip_bf16.h>

typedef __attribute__((ext_vector_type(16))) _Float16 v16h;
typedef __attribute__((ext_vector_type(8)))  _Float16 v8h;
typedef __attribute__((ext_vector_type(8)))  float    v8f;

// ---------------------------------------------------------------------------
// WMMA GEMM: C[M,Nc] = A[M,K] (f16, row-major) x Bt[Nc,K]^T (f16, row-major)
// + optional bias[Nc] + optional relu.
// One wave owns a 16x64 C strip (1 M-tile x 4 N-tiles): the A fragment is
// loaded once per K-step and reused by 4 WMMAs.  8 waves per block cover
// 8 M-tiles.  Requires K % 32 == 0, M % 16 == 0, Nc % 64 == 0.
// ---------------------------------------------------------------------------
__global__ __launch_bounds__(256) void wmma_gemm_f16(
    const _Float16* __restrict__ A, const _Float16* __restrict__ Bt,
    float* __restrict__ C, const float* __restrict__ bias,
    int M, int Nc, int K, int relu)
{
    const int wave = threadIdx.x >> 5;
    const int lane = threadIdx.x & 31;
    const int mtile = blockIdx.x * 8 + wave;
    if (mtile * 16 >= M) return;               // wave-uniform exit
    const int n0   = blockIdx.y * 64;          // 4 consecutive N-tiles
    const int half = lane >> 4;                // 0 or 1
    const int l15  = lane & 15;

    const _Float16* __restrict__ arow = A  + (size_t)(mtile * 16 + l15) * K;
    const _Float16* __restrict__ brow = Bt + (size_t)(n0 + l15) * K;
    const size_t bstep = (size_t)16 * K;       // row stride between N-tiles

    v8f acc[4] = {};
    for (int k = 0; k < K; k += 32) {
        // A 16x32 tile: lane l holds row M=l&15; elems 0..7 -> K = k+8*half+e,
        // elems 8..15 -> K = k+16+8*half+e  (ISA 7.12.2, 16-bit A layout)
        v8h alo = *(const v8h*)(arow + k + half * 8);
        v8h ahi = *(const v8h*)(arow + k + 16 + half * 8);
        v16h a;
#pragma unroll
        for (int e = 0; e < 8; ++e) { a[e] = alo[e]; a[e + 8] = ahi[e]; }
#pragma unroll
        for (int j = 0; j < 4; ++j) {
            // B 32x16 tile: lane l holds col N=l&15; elems 0..7 -> K = k+8*half+e,
            // elems 8..15 -> K = k+16+8*half+e
            const _Float16* bp = brow + (size_t)j * bstep + k + half * 8;
            v8h blo = *(const v8h*)(bp);
            v8h bhi = *(const v8h*)(bp + 16);
            v16h b;
#pragma unroll
            for (int e = 0; e < 8; ++e) { b[e] = blo[e]; b[e + 8] = bhi[e]; }
            acc[j] = __builtin_amdgcn_wmma_f32_16x16x32_f16(
                false, a, false, b, (short)0, acc[j], false, false);
        }
    }

#pragma unroll
    for (int j = 0; j < 4; ++j) {
        const int n = n0 + j * 16 + l15;
        const float bv = bias ? bias[n] : 0.0f;
#pragma unroll
        for (int r = 0; r < 8; ++r) {
            float v = acc[j][r] + bv;
            if (relu) v = fmaxf(v, 0.0f);
            C[(size_t)(mtile * 16 + r + half * 8) * Nc + n] = v;  // C/D layout
        }
    }
}

// ---------------------------------------------------------------------------
// Small utility kernels
// ---------------------------------------------------------------------------
__global__ void k_cvt_f16(const float* __restrict__ in, _Float16* __restrict__ out, int n)
{
    int i = blockIdx.x * 256 + threadIdx.x;
    if (i < n) out[i] = (_Float16)in[i];
}

// out[Nc][K] = (f16) in[K][Nc]   (transpose + convert, for B operands)
__global__ void k_transpose_cvt(const float* __restrict__ in, _Float16* __restrict__ out,
                                int K, int Nc)
{
    int i = blockIdx.x * 256 + threadIdx.x;
    if (i >= K * Nc) return;
    int n = i / K, k = i % K;
    out[i] = (_Float16)in[(size_t)k * Nc + n];
}

__global__ void k_deg_init(float* deg, int n)
{
    int i = blockIdx.x * 256 + threadIdx.x;
    if (i < n) deg[i] = 1.0f;                  // self loop
}

__global__ void k_deg_count(const int* __restrict__ dst, float* deg, int E)
{
    int i = blockIdx.x * 256 + threadIdx.x;
    if (i < E) atomicAdd(&deg[dst[i]], 1.0f);
}

__global__ void k_deg_rsqrt(float* deg, int n)
{
    int i = blockIdx.x * 256 + threadIdx.x;
    if (i < n) deg[i] = rsqrtf(deg[i]);        // deg >= 1 always
}

// agg[i][f] = lin[i][f] * dinv[i]^2   (self-loop term, also initializes agg)
__global__ void k_selfloop(const float* __restrict__ lin, const float* __restrict__ dinv,
                           float* __restrict__ agg, int n, int F)
{
    int i = blockIdx.x * 256 + threadIdx.x;
    if (i >= n * F) return;
    float d = dinv[i / F];
    agg[i] = lin[i] * d * d;
}

// agg[dst] += lin[src] * dinv[src]*dinv[dst], 4 floats per thread
__global__ void k_edge_scatter(const int* __restrict__ src, const int* __restrict__ dst,
                               const float* __restrict__ dinv, const float* __restrict__ lin,
                               float* __restrict__ agg, int E, int F)
{
    int chunks = F >> 2;
    int idx = blockIdx.x * 256 + threadIdx.x;
    if (idx >= E * chunks) return;
    int e = idx / chunks;
    int f = (idx % chunks) << 2;
    int s = src[e], d = dst[e];
    float w = dinv[s] * dinv[d];
    const float* hp = lin + (size_t)s * F + f;
    float* ap = agg + (size_t)d * F + f;
    atomicAdd(ap + 0, hp[0] * w);
    atomicAdd(ap + 1, hp[1] * w);
    atomicAdd(ap + 2, hp[2] * w);
    atomicAdd(ap + 3, hp[3] * w);
}

// h16 = (f16) relu(agg + b)
__global__ void k_bias_relu_cvt(const float* __restrict__ agg, const float* __restrict__ b,
                                _Float16* __restrict__ out, int n, int F)
{
    int i = blockIdx.x * 256 + threadIdx.x;
    if (i >= n * F) return;
    out[i] = (_Float16)fmaxf(agg[i] + b[i % F], 0.0f);
}

__global__ void k_bias_add(float* agg, const float* __restrict__ b, int n, int F)
{
    int i = blockIdx.x * 256 + threadIdx.x;
    if (i < n * F) agg[i] += b[i % F];
}

// g[i] = pb + h1[i,:] . pw
__global__ void k_score(const float* __restrict__ h1, const float* __restrict__ pw,
                        const float* __restrict__ pb, float* __restrict__ g, int n)
{
    int i = blockIdx.x * 256 + threadIdx.x;
    if (i >= n) return;
    float acc = pb[0];
    const float* hp = h1 + (size_t)i * 64;
#pragma unroll 8
    for (int c = 0; c < 64; ++c) acc += hp[c] * pw[c];
    g[i] = acc;
}

// Stable ascending rank (== inverse permutation of jnp.argsort, stable ties)
__global__ void k_rank(const float* __restrict__ g, int* __restrict__ rank, int n)
{
    __shared__ float sg[256];
    int i = blockIdx.x * 256 + threadIdx.x;
    float gi = (i < n) ? g[i] : 0.0f;
    int cnt = 0;
    for (int base = 0; base < n; base += 256) {
        int j = base + threadIdx.x;
        sg[threadIdx.x] = (j < n) ? g[j] : 0.0f;
        __syncthreads();
        int lim = min(256, n - base);
        for (int t = 0; t < lim; ++t) {
            float gj = sg[t];
            int jj = base + t;
            cnt += ((gj < gi) || (gj == gi && jj < i)) ? 1 : 0;
        }
        __syncthreads();
    }
    if (i < n) rank[i] = cnt;
}

// time-major sorted gated features: sx[rank[i]][c] = g[i] * h1[i][c]
__global__ void k_build_sx(const float* __restrict__ h1, const float* __restrict__ g,
                           const int* __restrict__ rank, _Float16* __restrict__ sx, int n)
{
    int idx = blockIdx.x * 256 + threadIdx.x;
    if (idx >= n * 64) return;
    int i = idx >> 6, c = idx & 63;
    sx[(size_t)rank[i] * 64 + c] = (_Float16)(g[i] * h1[idx]);
}

// col[t][c*5+k] = in[t+k-2][c]  (input time-major [t][64], zero-padded)
__global__ void k_im2col(const _Float16* __restrict__ in, _Float16* __restrict__ col, int n)
{
    int idx = blockIdx.x * 256 + threadIdx.x;
    if (idx >= n * 320) return;
    int t = idx / 320, ck = idx % 320;
    int c = ck / 5, k = ck % 5;
    int ts = t + k - 2;
    col[idx] = (ts >= 0 && ts < n) ? in[(size_t)ts * 64 + c] : (_Float16)0.0f;
}

// A2[i] = [ h1[i] (64) | conv2out[rank[i]] (64) ]  as f16
__global__ void k_build_a2(const float* __restrict__ h1, const float* __restrict__ c2,
                           const int* __restrict__ rank, _Float16* __restrict__ a2, int n)
{
    int idx = blockIdx.x * 256 + threadIdx.x;
    if (idx >= n * 128) return;
    int i = idx >> 7, c = idx & 127;
    float v = (c < 64) ? h1[(size_t)i * 64 + c]
                       : c2[(size_t)rank[i] * 64 + (c - 64)];
    a2[idx] = (_Float16)v;
}

// ---------------------------------------------------------------------------
// Host launch
// ---------------------------------------------------------------------------
static inline int cdiv(int a, int b) { return (a + b - 1) / b; }

extern "C" void kernel_launch(void* const* d_in, const int* in_sizes, int n_in,
                              void* d_out, int out_size, void* d_ws, size_t ws_size,
                              hipStream_t stream)
{
    const float* x   = (const float*)d_in[0];
    const int*   ei  = (const int*)  d_in[1];
    const float* w1  = (const float*)d_in[2];
    const float* b1  = (const float*)d_in[3];
    const float* w2  = (const float*)d_in[4];
    const float* b2  = (const float*)d_in[5];
    const float* pw  = (const float*)d_in[6];
    const float* pb  = (const float*)d_in[7];
    const float* cw1 = (const float*)d_in[8];
    const float* cb1 = (const float*)d_in[9];
    const float* cw2 = (const float*)d_in[10];
    const float* cb2 = (const float*)d_in[11];
    const float* lw  = (const float*)d_in[12];
    const float* lb  = (const float*)d_in[13];
    float* out = (float*)d_out;

    const int FIN = 256, HID = 256, OUT = 64;
    const int N = in_sizes[0] / FIN;        // 50000 (multiple of 16)
    const int E = in_sizes[1] / 2;          // 400000
    const int* src = ei;
    const int* dst = ei + E;

    // ---- workspace layout (256B-aligned bump allocator) ----
    char* ws = (char*)d_ws;
    size_t off = 0;
    auto alloc = [&](size_t bytes) -> char* {
        char* p = ws + off;
        off += (bytes + 255) & ~(size_t)255;
        return p;
    };
    _Float16* xf16  = (_Float16*)alloc((size_t)N * FIN * 2);
    _Float16* w1t   = (_Float16*)alloc((size_t)HID * FIN * 2);
    _Float16* w2t   = (_Float16*)alloc((size_t)OUT * HID * 2);
    _Float16* cw1t  = (_Float16*)alloc((size_t)OUT * OUT * 5 * 2);
    _Float16* cw2t  = (_Float16*)alloc((size_t)OUT * OUT * 5 * 2);
    _Float16* lwt   = (_Float16*)alloc((size_t)OUT * 2 * OUT * 2);
    _Float16* h16   = (_Float16*)alloc((size_t)N * HID * 2);
    float*    lin2  = (float*)   alloc((size_t)N * OUT * 4);
    float*    h1    = (float*)   alloc((size_t)N * OUT * 4);   // agg2 -> h1
    float*    dinv  = (float*)   alloc((size_t)N * 4);
    float*    gbuf  = (float*)   alloc((size_t)N * 4);
    int*      rank  = (int*)     alloc((size_t)N * 4);
    _Float16* sx16  = (_Float16*)alloc((size_t)N * OUT * 2);
    // region A: reused across phases
    char* regA = alloc((size_t)N * FIN * 4 * 2);               // 102.4 MB
    float*    lin1    = (float*)   regA;                       // [N,256] f32
    float*    agg1    = (float*)  (regA + (size_t)N * FIN * 4);
    _Float16* col     = (_Float16*)regA;                       // [N,320] f16 (after GCN1 dead)
    float*    c1out   = (float*)  (regA + (size_t)N * 320 * 2);
    _Float16* c1out16 = (_Float16*)((char*)c1out + (size_t)N * OUT * 4);
    float*    c2out   = (float*)  ((char*)c1out16 + (size_t)N * OUT * 2);
    _Float16* a2      = (_Float16*)((char*)c2out + (size_t)N * OUT * 4);
    (void)ws_size; (void)n_in; (void)out_size;

    const int MT = N / 16;                  // 3125 M-tiles
    dim3 blk(256);

    // ---- weight prep ----
    k_cvt_f16<<<cdiv(N * FIN, 256), blk, 0, stream>>>(x, xf16, N * FIN);
    k_transpose_cvt<<<cdiv(FIN * HID, 256), blk, 0, stream>>>(w1, w1t, FIN, HID);
    k_transpose_cvt<<<cdiv(HID * OUT, 256), blk, 0, stream>>>(w2, w2t, HID, OUT);
    k_cvt_f16<<<cdiv(OUT * OUT * 5, 256), blk, 0, stream>>>(cw1, cw1t, OUT * OUT * 5);
    k_cvt_f16<<<cdiv(OUT * OUT * 5, 256), blk, 0, stream>>>(cw2, cw2t, OUT * OUT * 5);
    k_transpose_cvt<<<cdiv(2 * OUT * OUT, 256), blk, 0, stream>>>(lw, lwt, 2 * OUT, OUT);

    // ---- degrees / norm ----
    k_deg_init<<<cdiv(N, 256), blk, 0, stream>>>(dinv, N);
    k_deg_count<<<cdiv(E, 256), blk, 0, stream>>>(dst, dinv, E);
    k_deg_rsqrt<<<cdiv(N, 256), blk, 0, stream>>>(dinv, N);

    // ---- GCN layer 1: lin1 = x @ W1 ; agg ; relu ; -> h16 ----
    wmma_gemm_f16<<<dim3(cdiv(MT, 8), HID / 64), blk, 0, stream>>>(
        xf16, w1t, lin1, nullptr, N, HID, FIN, 0);
    k_selfloop<<<cdiv(N * HID, 256), blk, 0, stream>>>(lin1, dinv, agg1, N, HID);
    k_edge_scatter<<<cdiv(E * (HID / 4), 256), blk, 0, stream>>>(
        src, dst, dinv, lin1, agg1, E, HID);
    k_bias_relu_cvt<<<cdiv(N * HID, 256), blk, 0, stream>>>(agg1, b1, h16, N, HID);

    // ---- GCN layer 2: lin2 = h @ W2 ; agg ; +b2 -> h1 ----
    wmma_gemm_f16<<<dim3(cdiv(MT, 8), OUT / 64), blk, 0, stream>>>(
        h16, w2t, lin2, nullptr, N, OUT, HID, 0);
    k_selfloop<<<cdiv(N * OUT, 256), blk, 0, stream>>>(lin2, dinv, h1, N, OUT);
    k_edge_scatter<<<cdiv(E * (OUT / 4), 256), blk, 0, stream>>>(
        src, dst, dinv, lin2, h1, E, OUT);
    k_bias_add<<<cdiv(N * OUT, 256), blk, 0, stream>>>(h1, b2, N, OUT);

    // ---- scores + stable rank (== inverse_idx) ----
    k_score<<<cdiv(N, 256), blk, 0, stream>>>(h1, pw, pb, gbuf, N);
    k_rank<<<cdiv(N, 256), blk, 0, stream>>>(gbuf, rank, N);

    // ---- sorted gated features, time-major f16 ----
    k_build_sx<<<cdiv(N * OUT, 256), blk, 0, stream>>>(h1, gbuf, rank, sx16, N);

    // ---- conv1 (im2col + WMMA GEMM, K=320) with relu ----
    k_im2col<<<cdiv(N * 320, 256), blk, 0, stream>>>(sx16, col, N);
    wmma_gemm_f16<<<dim3(cdiv(MT, 8), OUT / 64), blk, 0, stream>>>(
        col, cw1t, c1out, cb1, N, OUT, 320, 1);
    k_cvt_f16<<<cdiv(N * OUT, 256), blk, 0, stream>>>(c1out, c1out16, N * OUT);

    // ---- conv2 ----
    k_im2col<<<cdiv(N * 320, 256), blk, 0, stream>>>(c1out16, col, N);
    wmma_gemm_f16<<<dim3(cdiv(MT, 8), OUT / 64), blk, 0, stream>>>(
        col, cw2t, c2out, cb2, N, OUT, 320, 0);

    // ---- concat([h1, h2]) @ lw + lb -> out ----
    k_build_a2<<<cdiv(N * 128, 256), blk, 0, stream>>>(h1, c2out, rank, a2, N);
    wmma_gemm_f16<<<dim3(cdiv(MT, 8), OUT / 64), blk, 0, stream>>>(
        a2, lwt, out, lb, N, OUT, 2 * OUT, 0);
}
